// LE_ERNIE_BiGRU_GlobalPointer_Model_84241488543820
// MI455X (gfx1250) — matrix operand order repaired and checked by reading, not compiled, not run on hardware
//
#include <hip/hip_runtime.h>
#include <hip/hip_bf16.h>

// ---------------------------------------------------------------------------
// B=16, S=512, H=768, T=9, HD=64; dense out N = T*HD*2 = 1152
// ---------------------------------------------------------------------------
#define B_   16
#define S_   512
#define H_   768
#define T_   9
#define HD_  64
#define N1_  (T_ * HD_ * 2) /* 1152 */

typedef __attribute__((ext_vector_type(16))) __bf16 v16bf;
typedef __attribute__((ext_vector_type(8)))  float  v8f;
typedef __attribute__((ext_vector_type(4)))  unsigned su4;
typedef __attribute__((ext_vector_type(8)))  unsigned su8;

union Frag { unsigned int u[8]; v16bf v; };

static __device__ __forceinline__ unsigned short f2bf(float f) {
  union { float f; unsigned int u; } x; x.f = f;
  unsigned int u = x.u;
  return (unsigned short)((u + 0x7FFFu + ((u >> 16) & 1u)) >> 16);
}

// --- CDNA5 async memory->LDS path (ASYNCcnt-tracked, bypasses VGPRs) -------
static __device__ __forceinline__ unsigned lds_addr_of(const void* p) {
  // flat pointers to LDS carry the LDS byte offset in addr[31:0]
  return (unsigned)(unsigned long long)p;
}
static __device__ __forceinline__ void async_g2l_b128(unsigned lds_byte,
                                                      const void* g) {
  unsigned long long ga = (unsigned long long)g;
  asm volatile("global_load_async_to_lds_b128 %0, %1, off"
               :: "v"(lds_byte), "v"(ga) : "memory");
}
static __device__ __forceinline__ void wait_async0() {
  asm volatile("s_wait_asynccnt 0x0" ::: "memory");
}

// --- CDNA5 Tensor Data Mover: one descriptor moves a whole 2D tile ---------
// g1w0 packs: workgroup_mask=0 | data_size[17:16] | pad_enable[20] |
//             pad_interval[24:22] | pad_amount[31:25]
static __device__ __forceinline__ void tdm_load_2d(
    unsigned lds_byte, const void* gtile,
    unsigned tensor_d0, unsigned tensor_d1,   // tensor extents (elements)
    unsigned tile_d0, unsigned tile_d1,       // tile extents  (elements)
    unsigned stride_d0,                       // row stride    (elements)
    unsigned g1w0)
{
  unsigned long long ga = (unsigned long long)gtile;
  su4 g0;
  g0.x = 1u;                                   // count=1 (valid user D#)
  g0.y = lds_byte;                             // lds_addr
  g0.z = (unsigned)ga;                         // global_addr[31:0]
  g0.w = 0x80000000u | ((unsigned)(ga >> 32) & 0x01FFFFFFu); // [56:32]|type=2
  su8 g1;
  g1[0] = g1w0;
  g1[1] = (tensor_d0 & 0xFFFFu) << 16;                     // tensor_dim0.lo16
  g1[2] = (tensor_d0 >> 16) | ((tensor_d1 & 0xFFFFu) << 16);
  g1[3] = (tensor_d1 >> 16) | (tile_d0 << 16);             // tile_dim0
  g1[4] = tile_d1;                                         // tile_dim1 (d2=0)
  g1[5] = stride_d0;                                       // dim0 stride lo32
  g1[6] = 0u;
  g1[7] = 0u;
  asm volatile("tensor_load_to_lds %0, %1" :: "s"(g0), "s"(g1) : "memory");
}
// data_size=2B, pad 4 DWORDs after every 16 DWORDs (64B row -> 80B LDS row)
#define TDM_FLAGS_BF16_PAD80 0x06D10000u

// ---------------------------------------------------------------------------
// Kernel 0a: hidden f32 -> bf16 (streaming, packed stores)
// ---------------------------------------------------------------------------
__global__ __launch_bounds__(256) void le_cvt_hidden(
    const float* __restrict__ x, unsigned short* __restrict__ y)
{
  int i = (blockIdx.x * 256 + threadIdx.x) * 4; // total divisible by 1024
  float4 v = *(const float4*)(x + i);
  union { unsigned short s[4]; uint2 u; } p;
  p.s[0] = f2bf(v.x); p.s[1] = f2bf(v.y);
  p.s[2] = f2bf(v.z); p.s[3] = f2bf(v.w);
  *(uint2*)(y + i) = p.u;
}

// ---------------------------------------------------------------------------
// Kernel 0b: W [K=768][N=1152] f32 -> Wt [N][K] bf16 (LDS tile transpose)
// ---------------------------------------------------------------------------
__global__ __launch_bounds__(256) void le_transpose_w(
    const float* __restrict__ W, unsigned short* __restrict__ Wt)
{
  __shared__ unsigned short tile[32][33];
  int k0 = blockIdx.y * 32, n0 = blockIdx.x * 32;
  int c  = threadIdx.x & 31, r0 = threadIdx.x >> 5; // r0: 0..7
  #pragma unroll
  for (int i = 0; i < 4; ++i) {
    int r = r0 + 8 * i;
    tile[r][c] = f2bf(W[(size_t)(k0 + r) * N1_ + (n0 + c)]);
  }
  __syncthreads();
  #pragma unroll
  for (int i = 0; i < 4; ++i) {
    int r = r0 + 8 * i;
    Wt[(size_t)(n0 + r) * H_ + (k0 + c)] = tile[c][r];
  }
}

// ---------------------------------------------------------------------------
// Kernel 1: dense = hidden @ W + bias  (bf16 WMMA, TDM tile fills)
// Tile: BM=128 BN=128 BK=32, 256 threads = 8 waves (2x4), wave patch 64x32.
// Wave 0 issues one tensor_load_to_lds per tile; TDM pads each 64B row to
// the 80B LDS stride. All other waves just wait at the barrier.
// ---------------------------------------------------------------------------
__global__ __launch_bounds__(256) void le_dense_gemm_bf16wmma(
    const unsigned short* __restrict__ hA, const unsigned short* __restrict__ Wt,
    const float* __restrict__ bias, float* __restrict__ out)
{
  __shared__ __align__(16) unsigned short As[128 * 40]; // [m][k], 80B rows
  __shared__ __align__(16) unsigned short Bs[128 * 40]; // [n][k], 80B rows

  const int tid   = threadIdx.x;
  const int lane  = tid & 31;
  const int wave  = tid >> 5;
  const int waveM = wave >> 2;   // 0..1  -> 64 rows each
  const int waveN = wave & 3;    // 0..3  -> 32 cols each
  const int tileM = blockIdx.y * 128;
  const int tileN = blockIdx.x * 128;
  const unsigned asb = lds_addr_of(As);
  const unsigned bsb = lds_addr_of(Bs);

  v8f acc[4][2];
  #pragma unroll
  for (int i = 0; i < 4; ++i)
    #pragma unroll
    for (int j = 0; j < 2; ++j)
      #pragma unroll
      for (int e = 0; e < 8; ++e) acc[i][j][e] = 0.0f;

  for (int k0 = 0; k0 < H_; k0 += 32) {
    __syncthreads();                 // previous tiles fully consumed
    if (wave == 0) {
      tdm_load_2d(asb, &hA[(size_t)tileM * H_ + k0],
                  H_, B_ * S_, 32, 128, H_, TDM_FLAGS_BF16_PAD80);
      tdm_load_2d(bsb, &Wt[(size_t)tileN * H_ + k0],
                  H_, N1_, 32, 128, H_, TDM_FLAGS_BF16_PAD80);
      __builtin_amdgcn_s_wait_tensorcnt(0);
    }
    __syncthreads();                 // tiles visible to all waves

    Frag bfr[2];
    #pragma unroll
    for (int fn = 0; fn < 2; ++fn) {
      int n    = waveN * 32 + fn * 16 + (lane & 15);
      int kofs = (lane >= 16) ? 16 : 0;
      #pragma unroll
      for (int v = 0; v < 8; ++v)
        bfr[fn].u[v] = *(const unsigned int*)&Bs[n * 40 + kofs + 2 * v];
    }
    #pragma unroll
    for (int fm = 0; fm < 4; ++fm) {
      Frag afr;
      int m    = waveM * 64 + fm * 16 + (lane & 15);
      int kofs = (lane >= 16) ? 16 : 0;
      #pragma unroll
      for (int v = 0; v < 8; ++v) {
        int kk = 2 * v + kofs;
        afr.u[v] = *(const unsigned int*)&As[m * 40 + kk];
      }
      #pragma unroll
      for (int fn = 0; fn < 2; ++fn)
        acc[fm][fn] = __builtin_amdgcn_wmma_f32_16x16x32_bf16(
            false, afr.v, false, bfr[fn].v, (short)0, acc[fm][fn], false, false);
    }
  }

  // epilogue: D lane layout -> row = i + 8*(lane/16), col = lane%16
  #pragma unroll
  for (int fm = 0; fm < 4; ++fm)
    #pragma unroll
    for (int fn = 0; fn < 2; ++fn)
      #pragma unroll
      for (int i = 0; i < 8; ++i) {
        int m = tileM + waveM * 64 + fm * 16 + i + 8 * (lane >> 4);
        int n = tileN + waveN * 32 + fn * 16 + (lane & 15);
        out[(size_t)m * N1_ + n] = acc[fm][fn][i] + bias[n];
      }
}

// ---------------------------------------------------------------------------
// Kernel 2: RoPE. One thread per (b,s,t,j) pair; writes bf16 qw[B,T,S,HD]
// and transposed ktT[B,T,HD,S].
// ---------------------------------------------------------------------------
__global__ __launch_bounds__(256) void le_rope_split(
    const float* __restrict__ dense, unsigned short* __restrict__ qw,
    unsigned short* __restrict__ kt)
{
  int gid = blockIdx.x * 256 + threadIdx.x;
  int j = gid & 31;
  int t = (gid >> 5) % T_;
  int s = (gid / (32 * T_)) & (S_ - 1);
  int b = gid / (32 * T_ * S_);
  if (b >= B_) return;

  const float* p = dense + (size_t)(b * S_ + s) * N1_ + t * 128 + 4 * j;
  float q0 = p[0], k0 = p[1], q1 = p[2], k1 = p[3];

  // inv_freq = 10000^(-2j/64) = exp(-j * ln(1e4)/32)
  float inv_freq = __expf(-0.28782313662425572f * (float)j);
  float ang = (float)s * inv_freq;
  float sn, cs;
  __sincosf(ang, &sn, &cs);

  float qo0 = q0 * cs - q1 * sn, qo1 = q1 * cs + q0 * sn;
  float ko0 = k0 * cs - k1 * sn, ko1 = k1 * cs + k0 * sn;

  size_t qb = ((size_t)(b * T_ + t) * S_ + s) * HD_ + 2 * j;
  qw[qb]     = f2bf(qo0);
  qw[qb + 1] = f2bf(qo1);
  size_t kb = ((size_t)(b * T_ + t) * HD_ + 2 * j) * S_ + s;
  kt[kb]      = f2bf(ko0);
  kt[kb + S_] = f2bf(ko1);
}

// ---------------------------------------------------------------------------
// Kernel 3: logits[b,t] = (Q @ K^T)/8 - masks.  128x128 tile per block.
// Q tile filled via async-to-LDS (row stride 72 ushorts = 144 B = 9*16 B).
// ---------------------------------------------------------------------------
__global__ __launch_bounds__(256) void le_biaffine_scores(
    const unsigned short* __restrict__ qw, const unsigned short* __restrict__ kt,
    const int* __restrict__ am, float* __restrict__ out)
{
  __shared__ __align__(16) unsigned short Qs[128 * 72]; // [m][d]
  __shared__ __align__(16) unsigned short Ks[128 * 72]; // [n][d]

  const int tid   = threadIdx.x;
  const int lane  = tid & 31;
  const int wave  = tid >> 5;
  const int waveM = wave >> 2;
  const int waveN = wave & 3;
  const int bt = blockIdx.z;
  const int b  = bt / T_;
  const int m0 = blockIdx.y * 128;
  const int n0 = blockIdx.x * 128;

  const unsigned short* Q = qw + (size_t)bt * S_ * HD_;
  const unsigned short* K = kt + (size_t)bt * HD_ * S_;
  const unsigned qsb = lds_addr_of(Qs);

  // Q tile 128x64 bf16: 128 rows x 8 chunks(16B) = 1024 chunks, 4 per thread
  #pragma unroll
  for (int i = 0; i < 4; ++i) {
    int ch = tid * 4 + i;
    int r  = ch >> 3, c = ch & 7;
    async_g2l_b128(qsb + 2u * (unsigned)(r * 72 + c * 8),
                   &Q[(size_t)(m0 + r) * HD_ + c * 8]);
  }
  // K^T tile needs a transpose: K[d][n] -> Ks[n][d] (VGPR path)
  #pragma unroll
  for (int i = 0; i < 32; ++i) {
    int e = tid * 32 + i;
    int d = e >> 7, n = e & 127;
    Ks[n * 72 + d] = K[(size_t)d * S_ + (n0 + n)];
  }
  wait_async0();
  __syncthreads();

  v8f acc[4][2];
  #pragma unroll
  for (int i = 0; i < 4; ++i)
    #pragma unroll
    for (int j = 0; j < 2; ++j)
      #pragma unroll
      for (int e = 0; e < 8; ++e) acc[i][j][e] = 0.0f;

  #pragma unroll
  for (int kb = 0; kb < HD_; kb += 32) {
    Frag bfr[2];
    #pragma unroll
    for (int fn = 0; fn < 2; ++fn) {
      int n    = waveN * 32 + fn * 16 + (lane & 15);
      int kofs = kb + ((lane >= 16) ? 16 : 0);
      #pragma unroll
      for (int v = 0; v < 8; ++v)
        bfr[fn].u[v] = *(const unsigned int*)&Ks[n * 72 + kofs + 2 * v];
    }
    #pragma unroll
    for (int fm = 0; fm < 4; ++fm) {
      Frag afr;
      int m    = waveM * 64 + fm * 16 + (lane & 15);
      int kofs = kb + ((lane >= 16) ? 16 : 0);
      #pragma unroll
      for (int v = 0; v < 8; ++v) {
        int kk = 2 * v + kofs;
        afr.u[v] = *(const unsigned int*)&Qs[m * 72 + kk];
      }
      #pragma unroll
      for (int fn = 0; fn < 2; ++fn)
        acc[fm][fn] = __builtin_amdgcn_wmma_f32_16x16x32_bf16(
            false, afr.v, false, bfr[fn].v, (short)0, acc[fm][fn], false, false);
    }
  }

  const int* amb = am + b * S_;
  #pragma unroll
  for (int fm = 0; fm < 4; ++fm)
    #pragma unroll
    for (int fn = 0; fn < 2; ++fn) {
      int n   = n0 + waveN * 32 + fn * 16 + (lane & 15);
      float an = (float)amb[n];
      #pragma unroll
      for (int i = 0; i < 8; ++i) {
        int m = m0 + waveM * 64 + fm * 16 + i + 8 * (lane >> 4);
        float v = acc[fm][fn][i] * 0.125f;            // 1/sqrt(64)
        float pad = (float)amb[m] * an;
        v -= (1.0f - pad) * 1e12f;                    // padding mask
        if (m > n) v -= 1e12f;                        // triu (k=0) mask
        out[((size_t)bt * S_ + m) * S_ + n] = v;
      }
    }
}

// ---------------------------------------------------------------------------
extern "C" void kernel_launch(void* const* d_in, const int* in_sizes, int n_in,
                              void* d_out, int out_size, void* d_ws, size_t ws_size,
                              hipStream_t stream) {
  const float* hidden = (const float*)d_in[0];
  const int*   amask  = (const int*)d_in[1];
  const float* W      = (const float*)d_in[2];
  const float* bias   = (const float*)d_in[3];
  float*       outp   = (float*)d_out;

  // ws layout: dense f32 | qw bf16 | ktT bf16 | hA bf16 | Wt bf16
  char* ws = (char*)d_ws;
  float* dense = (float*)ws;
  ws += (size_t)B_ * S_ * N1_ * sizeof(float);
  unsigned short* qw = (unsigned short*)ws;
  ws += (size_t)B_ * T_ * S_ * HD_ * sizeof(unsigned short);
  unsigned short* kt = (unsigned short*)ws;
  ws += (size_t)B_ * T_ * S_ * HD_ * sizeof(unsigned short);
  unsigned short* hA = (unsigned short*)ws;
  ws += (size_t)B_ * S_ * H_ * sizeof(unsigned short);
  unsigned short* Wt = (unsigned short*)ws;

  (void)in_sizes; (void)n_in; (void)out_size; (void)ws_size;

  le_cvt_hidden<<<(B_ * S_ * H_) / 1024, 256, 0, stream>>>(hidden, hA);
  le_transpose_w<<<dim3(N1_ / 32, H_ / 32), 256, 0, stream>>>(W, Wt);

  le_dense_gemm_bf16wmma<<<dim3(N1_ / 128, (B_ * S_) / 128), 256, 0, stream>>>(
      hA, Wt, bias, dense);

  int ropeThreads = B_ * S_ * T_ * (HD_ / 2);
  le_rope_split<<<ropeThreads / 256, 256, 0, stream>>>(dense, qw, kt);

  le_biaffine_scores<<<dim3(S_ / 128, S_ / 128, B_ * T_), 256, 0, stream>>>(
      qw, kt, amask, outp);
}
